// CausalSelfAttention_60885456388908
// MI455X (gfx1250) — matrix operand order repaired and checked by reading, not compiled, not run on hardware
//
#include <hip/hip_runtime.h>
#include <hip/hip_bf16.h>

#define BB 2
#define TT 2048
#define CC 1024
#define HH 16
#define HD 64

typedef __attribute__((ext_vector_type(16))) __bf16 bf16x16;
typedef __attribute__((ext_vector_type(8)))  float  f32x8;

union Frag {
    bf16x16 v;
    uint4   q[2];
};

__device__ __forceinline__ unsigned short f2bf(float f) {
    union { float f; unsigned u; } c; c.f = f;
    unsigned u = c.u;
    u = (u + 0x7FFFu + ((u >> 16) & 1u)) >> 16;  // round-to-nearest-even
    return (unsigned short)u;
}

__device__ __forceinline__ f32x8 zero8() {
    f32x8 z = {0.f,0.f,0.f,0.f,0.f,0.f,0.f,0.f};
    return z;
}

// low 32 bits of a generic pointer to __shared__ == LDS byte offset
__device__ __forceinline__ unsigned lds_addr(const void* p) {
    return (unsigned)(unsigned long long)p;
}

// async copy 16B global -> LDS (per-lane), tracked by ASYNCcnt.
// No "memory" clobber: __syncthreads() fences are the ordering points.
__device__ __forceinline__ void async_b128(unsigned lds, const void* gptr) {
    asm volatile("global_load_async_to_lds_b128 %0, %1, off"
                 :: "v"(lds), "v"((unsigned long long)gptr));
}

#define WAIT_ASYNC(n) asm volatile("s_wait_asynccnt " #n ::: "memory")

// ---------------------------------------------------------------------------
// Convert / transpose kernels
// ---------------------------------------------------------------------------
__global__ void cvt_f32_to_bf16(const float* __restrict__ src,
                                unsigned short* __restrict__ dst, int n) {
    int i = blockIdx.x * 256 + threadIdx.x;
    if (i < n) dst[i] = f2bf(src[i]);
}

// src[K][N] (row-major) -> dst[N][K] bf16
__global__ void transpose_cvt(const float* __restrict__ src,
                              unsigned short* __restrict__ dst, int K, int N) {
    int i = blockIdx.x * 256 + threadIdx.x;
    if (i < K * N) {
        int k = i / N, n = i % N;
        dst[(size_t)n * K + k] = f2bf(src[i]);
    }
}

// ---------------------------------------------------------------------------
// WMMA GEMM core, double-buffered async global->LDS, branch-free pipeline.
// C[128x128] tile = A[M,K] * Bt[N,K]^T, bf16 in, f32 acc.
// 256 threads = 8 waves, wave (waveM 0..3, waveN 0..1) -> 32x64 subtile.
// sA/sB: 2 x 128x32 bf16 buffers each (8192 B per buffer).
// ---------------------------------------------------------------------------
__device__ __forceinline__ void gemm_core(
    const unsigned short* __restrict__ A, int lda,
    const unsigned short* __restrict__ Bt, int ldb,
    int K, int mBase, int nBase,
    unsigned short* sA, unsigned short* sB,
    f32x8 acc[8])
{
    const int tid   = threadIdx.x;
    const int wave  = tid >> 5;
    const int lane  = tid & 31;
    const int waveM = wave & 3;
    const int waveN = wave >> 2;
    const int l16   = lane & 15;
    const int lhalf = lane >> 4;

    #pragma unroll
    for (int i = 0; i < 8; i++) acc[i] = zero8();

    // per-thread copy slots: rows r and r+64, 16B chunk c4 (4 async b128/stage)
    const int r  = tid >> 2;
    const int c4 = tid & 3;
    const unsigned short* aP0 = A  + (size_t)(mBase + r) * lda + c4 * 8;
    const unsigned short* aP1 = aP0 + (size_t)64 * lda;
    const unsigned short* bP0 = Bt + (size_t)(nBase + r) * ldb + c4 * 8;
    const unsigned short* bP1 = bP0 + (size_t)64 * ldb;
    const unsigned lA0 = lds_addr(sA + r * 32 + c4 * 8);
    const unsigned lB0 = lds_addr(sB + r * 32 + c4 * 8);

    const int NK = K >> 5;

    // prologue: stage 0 into buffer 0
    async_b128(lA0,        aP0);
    async_b128(lA0 + 4096, aP1);
    async_b128(lB0,        bP0);
    async_b128(lB0 + 4096, bP1);
    aP0 += 32; aP1 += 32; bP0 += 32; bP1 += 32;

    #pragma unroll 2
    for (int ki = 0; ki < NK; ki++) {
        const int cur = ki & 1;
        const unsigned o = (unsigned)(cur ^ 1) << 13;
        __syncthreads();                 // buffer cur^1 free for refill
        // issue stage min(ki+1, NK-1); final iteration re-issues last stage
        async_b128(lA0 + o,        aP0);
        async_b128(lA0 + o + 4096, aP1);
        async_b128(lB0 + o,        bP0);
        async_b128(lB0 + o + 4096, bP1);
        WAIT_ASYNC(4);                   // own stage-ki copies landed (in-order)
        __syncthreads();                 // all waves' stage-ki data visible

        const int step = (ki + 2 < NK) ? 32 : 0;   // branchless pointer walk
        aP0 += step; aP1 += step; bP0 += step; bP1 += step;

        const unsigned short* cA = sA + cur * 4096;
        const unsigned short* cB = sB + cur * 4096;
        Frag af[2], bf[4];
        #pragma unroll
        for (int mt = 0; mt < 2; mt++) {
            int row = waveM * 32 + mt * 16 + l16;
            int kk  = lhalf * 8;
            af[mt].q[0] = *((const uint4*)(cA + row * 32 + kk));
            af[mt].q[1] = *((const uint4*)(cA + row * 32 + kk + 16));
        }
        #pragma unroll
        for (int nt = 0; nt < 4; nt++) {
            int nrow = waveN * 64 + nt * 16 + l16;
            int kk   = lhalf * 8;
            bf[nt].q[0] = *((const uint4*)(cB + nrow * 32 + kk));
            bf[nt].q[1] = *((const uint4*)(cB + nrow * 32 + kk + 16));
        }
        #pragma unroll
        for (int mt = 0; mt < 2; mt++)
            #pragma unroll
            for (int nt = 0; nt < 4; nt++)
                acc[mt * 4 + nt] = __builtin_amdgcn_wmma_f32_16x16x32_bf16(
                    false, af[mt].v, false, bf[nt].v,
                    (short)0, acc[mt * 4 + nt], false, false);
    }
}

// ---------------------------------------------------------------------------
// QKV GEMM: [4096,1024] x [1024,3072] -> scatter q/k/v in attention layouts
// q,k: [b,h,t,d] bf16 ; vt: [b,h,d,t] bf16 (transposed for P*V)
// ---------------------------------------------------------------------------
__global__ void __launch_bounds__(256, 1)
qkv_gemm(const unsigned short* __restrict__ xbf,
         const unsigned short* __restrict__ wqkvT,
         unsigned short* __restrict__ qb,
         unsigned short* __restrict__ kb,
         unsigned short* __restrict__ vtb)
{
    __shared__ unsigned short sA[2 * 128 * 32];
    __shared__ unsigned short sB[2 * 128 * 32];
    const int mBase = blockIdx.x * 128;
    const int nBase = blockIdx.y * 128;
    f32x8 acc[8];
    gemm_core(xbf, CC, wqkvT, CC, CC, mBase, nBase, sA, sB, acc);

    const int tid = threadIdx.x, wave = tid >> 5, lane = tid & 31;
    const int waveM = wave & 3, waveN = wave >> 2;
    const int l16 = lane & 15, lhalf = lane >> 4;
    #pragma unroll
    for (int mt = 0; mt < 2; mt++)
        #pragma unroll
        for (int nt = 0; nt < 4; nt++) {
            f32x8 a = acc[mt * 4 + nt];
            #pragma unroll
            for (int i = 0; i < 8; i++) {
                int m = mBase + waveM * 32 + mt * 16 + lhalf * 8 + i;
                int n = nBase + waveN * 64 + nt * 16 + l16;
                int b = m >> 11, t = m & 2047;
                int which = n >> 10, c = n & 1023;
                int h = c >> 6, d = c & 63;
                unsigned short val = f2bf(a[i]);
                size_t bh = (size_t)(b * HH + h);
                if (which == 0)      qb[(bh * TT + t) * HD + d] = val;
                else if (which == 1) kb[(bh * TT + t) * HD + d] = val;
                else                 vtb[(bh * HD + d) * TT + t] = val;
            }
        }
}

// ---------------------------------------------------------------------------
// Flash attention: one (b,h), one 64-row q tile per workgroup; 128 thr = 4 waves
// wave w owns 16 q rows; online softmax in f32; WMMA for S=Q*K^T and O+=P*V^T.
// K/V tiles double-buffered via async global->LDS copies (branch-free).
// output ybf [b, t, h*64+d] bf16
// ---------------------------------------------------------------------------
__global__ void __launch_bounds__(128, 1)
flash_attn(const unsigned short* __restrict__ qb,
           const unsigned short* __restrict__ kb,
           const unsigned short* __restrict__ vtb,
           unsigned short* __restrict__ ybf)
{
    __shared__ unsigned short sQ[64 * 64];
    __shared__ unsigned short sK[2 * 64 * 64];
    __shared__ unsigned short sV[2 * 64 * 64];     // vt tile: [d][key]
    __shared__ unsigned short sP[4 * 16 * 64];     // per-wave P staging

    const int bh = blockIdx.x;       // 0..31
    const int qt = blockIdx.y;       // 0..31 (q tile of 64)
    const int tid = threadIdx.x;     // 0..127
    const int wave = tid >> 5;
    const int lane = tid & 31;
    const int l16 = lane & 15, lhalf = lane >> 4;
    const size_t base = (size_t)bh * TT * HD;

    // per-thread copy slots: rows r0+16*i (i=0..3), 16B chunk c
    const int r0 = tid >> 3;
    const int c  = tid & 7;
    const unsigned short* kP = kb  + base + (size_t)r0 * HD + c * 8;  // +16*HD per i
    const unsigned short* vP = vtb + base + (size_t)r0 * TT + c * 8;  // +16*TT per i
    const unsigned lK0 = lds_addr(sK + r0 * 64 + c * 8);              // +2048B per i
    const unsigned lV0 = lds_addr(sV + r0 * 64 + c * 8);
    const unsigned lQ0 = lds_addr(sQ + r0 * 64 + c * 8);

    // prologue: Q tile (4 copies) then K/V stage 0 (8 copies)
    #pragma unroll
    for (int i = 0; i < 4; i++)
        async_b128(lQ0 + i * 2048,
                   qb + base + (size_t)(qt * 64 + r0 + 16 * i) * HD + c * 8);
    #pragma unroll
    for (int i = 0; i < 4; i++) {
        async_b128(lK0 + i * 2048, kP + (size_t)(16 * i) * HD);
        async_b128(lV0 + i * 2048, vP + (size_t)(16 * i) * TT);
    }
    WAIT_ASYNC(8);                    // Q copies (first 4 of 12, in-order) done
    __syncthreads();

    Frag qfrag[2];
    #pragma unroll
    for (int ks = 0; ks < 2; ks++) {
        int row = wave * 16 + l16;
        int kk  = ks * 32 + lhalf * 8;
        qfrag[ks].q[0] = *((const uint4*)(sQ + row * 64 + kk));
        qfrag[ks].q[1] = *((const uint4*)(sQ + row * 64 + kk + 16));
    }

    f32x8 oacc[4];
    #pragma unroll
    for (int nt = 0; nt < 4; nt++) oacc[nt] = zero8();
    float mrow[8], lrow[8];
    #pragma unroll
    for (int i = 0; i < 8; i++) { mrow[i] = -__builtin_inff(); lrow[i] = 0.f; }

    const int qg = qt * 64 + wave * 16 + lhalf * 8;   // this lane's first q row

    for (int j = 0; j <= qt; j++) {
        const int cur = j & 1;
        const unsigned o = (unsigned)(cur ^ 1) << 13;
        __syncthreads();              // buffer cur^1 free for refill
        // issue stage min(j+1, qt); final iteration re-issues last stage
        const int jn = (j < qt) ? j + 1 : qt;
        const size_t kj = (size_t)jn * 64 * HD;
        const size_t vj = (size_t)jn * 64;
        #pragma unroll
        for (int i = 0; i < 4; i++) {
            async_b128(lK0 + o + i * 2048, kP + kj + (size_t)(16 * i) * HD);
            async_b128(lV0 + o + i * 2048, vP + vj + (size_t)(16 * i) * TT);
        }
        WAIT_ASYNC(8);                // own stage-j copies landed (in-order)
        __syncthreads();              // all waves' stage-j K/V visible

        const unsigned short* cK = sK + cur * 4096;
        const unsigned short* cV = sV + cur * 4096;

        // S = (Q * K^T) * scale  -- 4 n-tiles of 16 keys each
        f32x8 sfrag[4];
        #pragma unroll
        for (int nt = 0; nt < 4; nt++) {
            f32x8 s = zero8();
            #pragma unroll
            for (int ks = 0; ks < 2; ks++) {
                Frag bk;
                int kr = nt * 16 + l16;
                int kk = ks * 32 + lhalf * 8;
                bk.q[0] = *((const uint4*)(cK + kr * 64 + kk));
                bk.q[1] = *((const uint4*)(cK + kr * 64 + kk + 16));
                s = __builtin_amdgcn_wmma_f32_16x16x32_bf16(
                        false, qfrag[ks].v, false, bk.v, (short)0, s, false, false);
            }
            // scale + causal mask (branchless: predicate false whenever j < qt)
            const int kg = j * 64 + nt * 16 + l16;
            #pragma unroll
            for (int i = 0; i < 8; i++) {
                float sv = s[i] * 0.125f;                 // 1/sqrt(64)
                s[i] = (kg > qg + i) ? -__builtin_inff() : sv;
            }
            sfrag[nt] = s;
        }

        // online softmax: row stats live across 16-lane half-wave groups
        float nmax[8], corr[8], rsum[8];
        #pragma unroll
        for (int i = 0; i < 8; i++) {
            float v = fmaxf(fmaxf(sfrag[0][i], sfrag[1][i]),
                            fmaxf(sfrag[2][i], sfrag[3][i]));
            #pragma unroll
            for (int off = 1; off < 16; off <<= 1)
                v = fmaxf(v, __shfl_xor(v, off, 32));
            nmax[i] = fmaxf(mrow[i], v);
            corr[i] = __expf(mrow[i] - nmax[i]);
            mrow[i] = nmax[i];
            rsum[i] = 0.f;
        }
        #pragma unroll
        for (int nt = 0; nt < 4; nt++)
            #pragma unroll
            for (int i = 0; i < 8; i++) {
                float p = __expf(sfrag[nt][i] - nmax[i]);
                sfrag[nt][i] = p;
                rsum[i] += p;
            }
        #pragma unroll
        for (int i = 0; i < 8; i++) {
            float v = rsum[i];
            #pragma unroll
            for (int off = 1; off < 16; off <<= 1)
                v += __shfl_xor(v, off, 32);
            lrow[i] = lrow[i] * corr[i] + v;
        }
        #pragma unroll
        for (int nt = 0; nt < 4; nt++)
            #pragma unroll
            for (int i = 0; i < 8; i++) oacc[nt][i] *= corr[i];

        // stage P (C-layout -> LDS -> A-layout), per-wave region
        #pragma unroll
        for (int nt = 0; nt < 4; nt++)
            #pragma unroll
            for (int i = 0; i < 8; i++)
                sP[wave * 1024 + (lhalf * 8 + i) * 64 + nt * 16 + l16] =
                    f2bf(sfrag[nt][i]);
        __syncthreads();

        // O += P * V^T
        #pragma unroll
        for (int ks = 0; ks < 2; ks++) {
            Frag pf;
            int kk = ks * 32 + lhalf * 8;
            pf.q[0] = *((const uint4*)(sP + wave * 1024 + l16 * 64 + kk));
            pf.q[1] = *((const uint4*)(sP + wave * 1024 + l16 * 64 + kk + 16));
            #pragma unroll
            for (int nt = 0; nt < 4; nt++) {
                Frag vf;
                int vr = nt * 16 + l16;
                int vk = ks * 32 + lhalf * 8;
                vf.q[0] = *((const uint4*)(cV + vr * 64 + vk));
                vf.q[1] = *((const uint4*)(cV + vr * 64 + vk + 16));
                oacc[nt] = __builtin_amdgcn_wmma_f32_16x16x32_bf16(
                               false, pf.v, false, vf.v, (short)0, oacc[nt],
                               false, false);
            }
        }
    }

    // normalize and write y [b, t, h*64+d]
    const int b = bh >> 4, h = bh & 15;
    #pragma unroll
    for (int nt = 0; nt < 4; nt++)
        #pragma unroll
        for (int i = 0; i < 8; i++) {
            float val = oacc[nt][i] / lrow[i];
            int t   = qt * 64 + wave * 16 + lhalf * 8 + i;
            int col = h * 64 + nt * 16 + l16;
            ybf[((size_t)b * TT + t) * CC + col] = f2bf(val);
        }
}

// ---------------------------------------------------------------------------
// Proj GEMM: [4096,1024] x [1024,1024] -> f32 output
// ---------------------------------------------------------------------------
__global__ void __launch_bounds__(256, 1)
proj_gemm(const unsigned short* __restrict__ ybf,
          const unsigned short* __restrict__ wprojT,
          float* __restrict__ out)
{
    __shared__ unsigned short sA[2 * 128 * 32];
    __shared__ unsigned short sB[2 * 128 * 32];
    const int mBase = blockIdx.x * 128;
    const int nBase = blockIdx.y * 128;
    f32x8 acc[8];
    gemm_core(ybf, CC, wprojT, CC, CC, mBase, nBase, sA, sB, acc);

    const int tid = threadIdx.x, wave = tid >> 5, lane = tid & 31;
    const int waveM = wave & 3, waveN = wave >> 2;
    const int l16 = lane & 15, lhalf = lane >> 4;
    #pragma unroll
    for (int mt = 0; mt < 2; mt++)
        #pragma unroll
        for (int nt = 0; nt < 4; nt++) {
            f32x8 a = acc[mt * 4 + nt];
            #pragma unroll
            for (int i = 0; i < 8; i++) {
                int m = mBase + waveM * 32 + mt * 16 + lhalf * 8 + i;
                int n = nBase + waveN * 64 + nt * 16 + l16;
                out[(size_t)m * CC + n] = a[i];
            }
        }
}

// ---------------------------------------------------------------------------
extern "C" void kernel_launch(void* const* d_in, const int* in_sizes, int n_in,
                              void* d_out, int out_size, void* d_ws, size_t ws_size,
                              hipStream_t stream) {
    const float* x      = (const float*)d_in[0];  // [2,2048,1024]
    const float* w_qkv  = (const float*)d_in[1];  // [1024,3072]
    const float* w_proj = (const float*)d_in[2];  // [1024,1024]
    float* out = (float*)d_out;                   // [2,2048,1024]

    const size_t NX  = (size_t)BB * TT * CC;      // 4194304
    const size_t NWQ = (size_t)CC * 3 * CC;       // 3145728
    const size_t NWP = (size_t)CC * CC;           // 1048576
    const size_t NQ  = (size_t)BB * HH * TT * HD; // 4194304

    unsigned short* xbf    = (unsigned short*)d_ws;
    unsigned short* wqkvT  = xbf + NX;
    unsigned short* wprojT = wqkvT + NWQ;
    unsigned short* qb     = wprojT + NWP;
    unsigned short* kb     = qb + NQ;
    unsigned short* vtb    = kb + NQ;
    unsigned short* ybf    = vtb + NQ;

    cvt_f32_to_bf16<<<(int)((NX + 255) / 256), 256, 0, stream>>>(x, xbf, (int)NX);
    transpose_cvt<<<(int)((NWQ + 255) / 256), 256, 0, stream>>>(w_qkv, wqkvT, CC, 3 * CC);
    transpose_cvt<<<(int)((NWP + 255) / 256), 256, 0, stream>>>(w_proj, wprojT, CC, CC);

    qkv_gemm<<<dim3(32, 24), 256, 0, stream>>>(xbf, wqkvT, qb, kb, vtb);
    flash_attn<<<dim3(BB * HH, TT / 64), 128, 0, stream>>>(qb, kb, vtb, ybf);
    proj_gemm<<<dim3(32, 8), 256, 0, stream>>>(ybf, wprojT, out);
}
